// SwiGLUMLP_9500467659038
// MI455X (gfx1250) — compile-verified, matrix operand order checked by reading, not run on hardware
//
#include <hip/hip_runtime.h>
#include <hip/hip_bf16.h>

// ---------------------------------------------------------------------------
// Octonion ternary SwiGLU MLP for MI455X (gfx1250), wave32 + WMMA bf16 + TDM.
// Pipeline: absmean scales -> pack bf16 signed-permuted weights (transposed,
// zero-padded to 10944) -> fused gate/up GEMM + SwiGLU -> down GEMM.
// GEMM tiles are staged into LDS by the Tensor Data Mover (tensor_load_to_lds,
// double-buffered, TENSORcnt-tracked) instead of VGPR round-trips.
// ---------------------------------------------------------------------------

typedef __bf16 v16bf __attribute__((ext_vector_type(16)));
typedef __bf16 v8bf  __attribute__((ext_vector_type(8)));
typedef float  v8f   __attribute__((ext_vector_type(8)));
typedef int    v4i   __attribute__((ext_vector_type(4)));
typedef int    v8i   __attribute__((ext_vector_type(8)));

constexpr int D_EMB = 4096;    // N_EMBD
constexpr int HID   = 10928;   // HIDDEN
constexpr int HPAD  = 10944;   // = 32*342 = 64*171 (pad for clean K/N tiling)
constexpr int BT    = 4096;    // 2*2048 tokens
constexpr int D8    = 512;     // D_EMB/8
constexpr int H8    = 1366;    // HID/8

// ---- octonion structure constants: unique (j, sign) per (i,k) --------------
struct OctTab { int j[64]; float s[64]; };

constexpr OctTab make_oct() {
    float S[8][8][8] = {};
    S[0][0][0] = 1.0f;
    for (int i = 1; i < 8; ++i) { S[0][i][i] = 1.0f; S[i][0][i] = 1.0f; S[i][i][0] = -1.0f; }
    const int T[7][3] = {{1,2,3},{1,4,5},{1,7,6},{2,4,6},{2,5,7},{3,4,7},{3,6,5}};
    for (int t = 0; t < 7; ++t) {
        int a = T[t][0], b = T[t][1], c = T[t][2];
        int P[3][3] = {{a,b,c},{b,c,a},{c,a,b}};
        for (int p = 0; p < 3; ++p) {
            S[P[p][0]][P[p][1]][P[p][2]] = 1.0f;
            S[P[p][1]][P[p][0]][P[p][2]] = -1.0f;
        }
    }
    OctTab tab{};
    for (int i = 0; i < 8; ++i)
        for (int k = 0; k < 8; ++k)
            for (int j = 0; j < 8; ++j)
                if (S[j][i][k] != 0.0f) { tab.j[i*8+k] = j; tab.s[i*8+k] = S[j][i][k]; }
    return tab;
}

__constant__ OctTab OCT = make_oct();

// ---- helpers ---------------------------------------------------------------
__device__ inline unsigned short f32_to_bf16(float f) {
    unsigned int u = __float_as_uint(f);
    u += 0x7FFFu + ((u >> 16) & 1u);      // round-to-nearest-even
    return (unsigned short)(u >> 16);
}

// Generic pointers into LDS carry the LDS byte offset in their low 32 bits
// (ISA 10.2, LDS aperture).
__device__ inline unsigned lds_off(const void* p) {
    return (unsigned)(unsigned long long)(uintptr_t)p;
}

// Issue a TDM 2D tile load: tile_x (contiguous, elems) x tile_y rows of 2-byte
// elements from a row-major matrix with row pitch `stride` elems, into LDS at
// byte offset `lds_addr` (rows packed contiguously).  D# per ISA ch.8:
//   group0: count=1 | lds_addr | global_addr[56:0] | type=2
//   group1: data_size=1(2B), tensor dims == tile dims (tiles always in-bounds),
//           tile_dim0/1, tensor_dim0_stride = row pitch.
__device__ inline void tdm_load_tile_2d(unsigned lds_addr, const void* gptr,
                                        int tile_x, int tile_y, int stride) {
    unsigned long long ga = (unsigned long long)(uintptr_t)gptr;
    unsigned long long st0 = (unsigned long long)(unsigned)stride;
    unsigned td0 = (unsigned)tile_x, td1 = (unsigned)tile_y;
    v4i g0;
    g0[0] = 1;                                                   // count=1
    g0[1] = (int)lds_addr;                                       // lds_addr
    g0[2] = (int)(unsigned)(ga & 0xFFFFFFFFu);                   // gaddr[31:0]
    g0[3] = (int)(((unsigned)(ga >> 32) & 0x01FFFFFFu) | (2u << 30)); // [56:32]|type=2
    v8i g1;
    g1[0] = 0x00010000;                                          // data_size=1 (2B)
    g1[1] = (int)((td0 & 0xFFFFu) << 16);                        // tensor_dim0[15:0]
    g1[2] = (int)((td0 >> 16) | ((td1 & 0xFFFFu) << 16));        // td0[31:16]|td1[15:0]
    g1[3] = (int)((td1 >> 16) | (td0 << 16));                    // td1[31:16]|tile_dim0
    g1[4] = (int)(td1 & 0xFFFFu);                                // tile_dim1 (tile_dim2=0)
    g1[5] = (int)(unsigned)(st0 & 0xFFFFFFFFu);                  // dim0_stride[31:0]
    g1[6] = (int)(unsigned)((st0 >> 32) & 0xFFFFu);              // dim0_stride[47:32]
    g1[7] = 0;
    asm volatile("tensor_load_to_lds %0, %1" :: "s"(g0), "s"(g1) : "memory");
}

// ---- scale reduction (deterministic two-pass, no float atomics) ------------
__global__ void absmean_partial(const float* __restrict__ w0,
                                const float* __restrict__ w1,
                                const float* __restrict__ w2,
                                int n, float* __restrict__ partials) {
    const float* w = (blockIdx.y == 0) ? w0 : (blockIdx.y == 1 ? w1 : w2);
    __shared__ float red[256];
    float s = 0.0f;
    for (int i = blockIdx.x * 256 + threadIdx.x; i < n; i += 256 * gridDim.x)
        s += fabsf(w[i]);
    red[threadIdx.x] = s;
    __syncthreads();
    for (int off = 128; off > 0; off >>= 1) {
        if ((int)threadIdx.x < off) red[threadIdx.x] += red[threadIdx.x + off];
        __syncthreads();
    }
    if (threadIdx.x == 0) partials[blockIdx.y * 256 + blockIdx.x] = red[0];
}

__global__ void absmean_final(const float* __restrict__ partials,
                              float* __restrict__ scales, int n) {
    __shared__ float red[256];
    red[threadIdx.x] = partials[blockIdx.x * 256 + threadIdx.x];
    __syncthreads();
    for (int off = 128; off > 0; off >>= 1) {
        if ((int)threadIdx.x < off) red[threadIdx.x] += red[threadIdx.x + off];
        __syncthreads();
    }
    if (threadIdx.x == 0) scales[blockIdx.x] = red[0] / (float)n;
}

// ---- pack x (f32 -> bf16) --------------------------------------------------
__global__ void pack_x(const float* __restrict__ x, unsigned short* __restrict__ xb, int n) {
    for (int i = blockIdx.x * blockDim.x + threadIdx.x; i < n; i += gridDim.x * blockDim.x)
        xb[i] = f32_to_bf16(x[i]);
}

// ---- pack gate/up effective matrix, transposed: Mt[h=0..HPAD)[dIn=0..D) ----
// M_eff[(i*512+d),(k*1366+o)] = sign(i,k) * tq(w[j(i,k), o, d])
__global__ void pack_gate_up(const float* __restrict__ w,
                             const float* __restrict__ scale_p,
                             unsigned short* __restrict__ Mt) {
    const float scale = *scale_p;
    const float inv = 1.0f / scale;
    const int total = HPAD * D_EMB;
    for (int idx = blockIdx.x * blockDim.x + threadIdx.x; idx < total;
         idx += gridDim.x * blockDim.x) {
        int dIn = idx & (D_EMB - 1);
        int h   = idx >> 12;
        unsigned short v = 0;
        if (h < HID) {
            int k = h / H8, o = h - k * H8;
            int i = dIn >> 9, d = dIn & 511;
            int j = OCT.j[i*8 + k];
            float sgn = OCT.s[i*8 + k];
            float wv = w[(j * H8 + o) * D8 + d];   // w_gate/up: (8, 1366, 512)
            float q = rintf(wv * inv);
            q = fminf(1.0f, fmaxf(-1.0f, q));
            v = f32_to_bf16(sgn * scale * q);
        }
        Mt[idx] = v;
    }
}

// ---- pack down effective matrix, transposed: Mt[dOut=0..D)[kk=0..HPAD) -----
// M_eff[(i*1366+d),(k*512+o)] = sign(i,k) * tq(w_down[j(i,k), o, d])
__global__ void pack_down(const float* __restrict__ w,
                          const float* __restrict__ scale_p,
                          unsigned short* __restrict__ Mt) {
    const float scale = *scale_p;
    const float inv = 1.0f / scale;
    const int total = D_EMB * HPAD;
    for (int idx = blockIdx.x * blockDim.x + threadIdx.x; idx < total;
         idx += gridDim.x * blockDim.x) {
        int kk = idx % HPAD;
        int n  = idx / HPAD;
        unsigned short v = 0;
        if (kk < HID) {
            int k = n >> 9, o = n & 511;
            int i = kk / H8, d = kk - i * H8;
            int j = OCT.j[i*8 + k];
            float sgn = OCT.s[i*8 + k];
            float wv = w[(j * D8 + o) * H8 + d];   // w_down: (8, 512, 1366)
            float q = rintf(wv * inv);
            q = fminf(1.0f, fmaxf(-1.0f, q));
            v = f32_to_bf16(sgn * scale * q);
        }
        Mt[idx] = v;
    }
}

// ---- WMMA GEMM with TDM-staged, double-buffered LDS tiles ------------------
// C[M,N] = A[M,K] * B^T[N,K]^T.  Block tile 128x64, 8 waves (4x2), each wave
// 32x32 = 2x2 v_wmma_f32_16x16x32_bf16 tiles.  FUSED: two B matrices (gate &
// up), SwiGLU epilogue -> bf16 Hout.  Else: f32 epilogue -> Cout.
// Wave 0 drives the TDM: per k-step it waits TENSORcnt for the current
// buffers, everyone barriers (which also drains DScnt, freeing the other
// buffer), then wave 0 issues next-k TDM loads while all waves run WMMA.
template <bool FUSED>
__global__ __launch_bounds__(256, 2)
void oct_gemm(const unsigned short* __restrict__ A,
              const unsigned short* __restrict__ B0,
              const unsigned short* __restrict__ B1,
              unsigned short* __restrict__ Hout,
              float* __restrict__ Cout,
              int K, int Astride, int Bstride, int OutStride) {
    __shared__ __align__(128) unsigned short As[2][128 * 32];            // 16 KB
    __shared__ __align__(128) unsigned short Bs0[2][64 * 32];            // 8 KB
    __shared__ __align__(128) unsigned short Bs1[2][FUSED ? 64 * 32 : 8];// 8 KB / tiny

    const int tid  = threadIdx.x;
    const int lane = tid & 31;
    const int wid  = tid >> 5;
    const int wm   = (wid & 3) * 32;       // wave row offset in block tile
    const int wn   = (wid >> 2) * 32;      // wave col offset in block tile
    const int m0   = blockIdx.y * 128;
    const int n0   = blockIdx.x * 64;

    const v8f vzero = {0.f, 0.f, 0.f, 0.f, 0.f, 0.f, 0.f, 0.f};
    v8f accG[2][2], accU[2][2];
#pragma unroll
    for (int mi = 0; mi < 2; ++mi)
#pragma unroll
        for (int ni = 0; ni < 2; ++ni) { accG[mi][ni] = vzero; accU[mi][ni] = vzero; }

    const int row16 = lane & 15;           // row (A) / col (B) within 16
    const int halfk = (lane >> 4) * 8;     // K sub-block select

    const unsigned short* Abase = A  + (size_t)m0 * Astride;
    const unsigned short* Bbase = B0 + (size_t)n0 * Bstride;
    const unsigned short* Cbase = FUSED ? (B1 + (size_t)n0 * Bstride) : (const unsigned short*)0;

    const int NK = K >> 5;                 // K / 32
    int cur = 0;

    if (wid == 0) {                        // prologue: fill buffer 0
        tdm_load_tile_2d(lds_off(&As[0][0]),  Abase, 32, 128, Astride);
        tdm_load_tile_2d(lds_off(&Bs0[0][0]), Bbase, 32, 64,  Bstride);
        if (FUSED)
            tdm_load_tile_2d(lds_off(&Bs1[0][0]), Cbase, 32, 64, Bstride);
    }

    for (int it = 0; it < NK; ++it) {
        if (wid == 0)
            __builtin_amdgcn_s_wait_tensorcnt(0);   // buffers[cur] landed in LDS
        __syncthreads();                            // visibility + DScnt drain

        if (wid == 0 && (it + 1) < NK) {            // prefetch next k into other buf
            int nb = cur ^ 1;
            int kn = (it + 1) << 5;
            tdm_load_tile_2d(lds_off(&As[nb][0]),  Abase + kn, 32, 128, Astride);
            tdm_load_tile_2d(lds_off(&Bs0[nb][0]), Bbase + kn, 32, 64,  Bstride);
            if (FUSED)
                tdm_load_tile_2d(lds_off(&Bs1[nb][0]), Cbase + kn, 32, 64, Bstride);
        }

        v16bf a[2], bg[2], bu[2];
#pragma unroll
        for (int mi = 0; mi < 2; ++mi) {
            const unsigned short* p = &As[cur][(wm + mi * 16 + row16) * 32];
            union { v16bf v; v8bf h[2]; } u;
            u.h[0] = *(const v8bf*)(p + halfk);
            u.h[1] = *(const v8bf*)(p + 16 + halfk);
            a[mi] = u.v;
        }
#pragma unroll
        for (int ni = 0; ni < 2; ++ni) {
            const unsigned short* p = &Bs0[cur][(wn + ni * 16 + row16) * 32];
            union { v16bf v; v8bf h[2]; } u;
            u.h[0] = *(const v8bf*)(p + halfk);
            u.h[1] = *(const v8bf*)(p + 16 + halfk);
            bg[ni] = u.v;
            if (FUSED) {
                const unsigned short* q = &Bs1[cur][(wn + ni * 16 + row16) * 32];
                union { v16bf v; v8bf h[2]; } u2;
                u2.h[0] = *(const v8bf*)(q + halfk);
                u2.h[1] = *(const v8bf*)(q + 16 + halfk);
                bu[ni] = u2.v;
            }
        }

#pragma unroll
        for (int mi = 0; mi < 2; ++mi)
#pragma unroll
            for (int ni = 0; ni < 2; ++ni) {
                accG[mi][ni] = __builtin_amdgcn_wmma_f32_16x16x32_bf16(
                    false, a[mi], false, bg[ni], (short)0, accG[mi][ni], false, false);
                if (FUSED)
                    accU[mi][ni] = __builtin_amdgcn_wmma_f32_16x16x32_bf16(
                        false, a[mi], false, bu[ni], (short)0, accU[mi][ni], false, false);
            }
        cur ^= 1;
    }

    // ---- epilogue: C/D layout (ISA 7.12.2): VGPR r -> row r (lanes 0-15) or
    //      row r+8 (lanes 16-31); col = lane%16.
    const int ncol  = lane & 15;
    const int rbase = (lane >> 4) * 8;
#pragma unroll
    for (int mi = 0; mi < 2; ++mi)
#pragma unroll
        for (int ni = 0; ni < 2; ++ni)
#pragma unroll
            for (int r = 0; r < 8; ++r) {
                int gm = m0 + wm + mi * 16 + rbase + r;
                int gn = n0 + wn + ni * 16 + ncol;
                if (FUSED) {
                    float g = accG[mi][ni][r];
                    float u = accU[mi][ni][r];
                    float h = (g / (1.0f + __expf(-g))) * u;   // silu(g)*u
                    Hout[gm * OutStride + gn] = f32_to_bf16(h);
                } else {
                    Cout[gm * OutStride + gn] = accG[mi][ni][r];
                }
            }
}

// ---- workspace layout ------------------------------------------------------
constexpr size_t SZ_XB = (size_t)BT * D_EMB * 2;     // 33,554,432
constexpr size_t SZ_MT = (size_t)HPAD * D_EMB * 2;   // 89,653,248
constexpr size_t O_XB = 4096;
constexpr size_t O_MG = O_XB + SZ_XB;
constexpr size_t O_MU = O_MG + SZ_MT;
constexpr size_t O_MD = O_MU + SZ_MT;
constexpr size_t O_HB = O_MD + SZ_MT;                // total ~374 MB

extern "C" void kernel_launch(void* const* d_in, const int* in_sizes, int n_in,
                              void* d_out, int out_size, void* d_ws, size_t ws_size,
                              hipStream_t stream) {
    const float* x  = (const float*)d_in[0];
    const float* wg = (const float*)d_in[1];
    const float* wu = (const float*)d_in[2];
    const float* wd = (const float*)d_in[3];
    float* out = (float*)d_out;

    char* ws = (char*)d_ws;
    float* partials = (float*)ws;                    // 3*256 f32
    float* scales   = (float*)(ws + 3072);           // 3 f32
    unsigned short* xb  = (unsigned short*)(ws + O_XB);
    unsigned short* MGt = (unsigned short*)(ws + O_MG);
    unsigned short* MUt = (unsigned short*)(ws + O_MU);
    unsigned short* MDt = (unsigned short*)(ws + O_MD);
    unsigned short* hb  = (unsigned short*)(ws + O_HB);

    const int nw = in_sizes[1];                      // 8*1366*512 (all three equal)

    absmean_partial<<<dim3(256, 3), 256, 0, stream>>>(wg, wu, wd, nw, partials);
    absmean_final<<<3, 256, 0, stream>>>(partials, scales, nw);

    pack_x<<<4096, 256, 0, stream>>>(x, xb, BT * D_EMB);
    pack_gate_up<<<8192, 256, 0, stream>>>(wg, scales + 0, MGt);
    pack_gate_up<<<8192, 256, 0, stream>>>(wu, scales + 1, MUt);
    pack_down<<<8192, 256, 0, stream>>>(wd, scales + 2, MDt);

    // GEMM1 (fused gate+up + SwiGLU): M=4096, N=10944, K=4096 -> hb (bf16)
    oct_gemm<true><<<dim3(HPAD / 64, BT / 128), 256, 0, stream>>>(
        xb, MGt, MUt, hb, nullptr, D_EMB, D_EMB, D_EMB, HPAD);

    // GEMM2 (down): M=4096, N=4096, K=10944 -> out (f32)
    oct_gemm<false><<<dim3(D_EMB / 64, BT / 128), 256, 0, stream>>>(
        hb, MDt, nullptr, nullptr, out, HPAD, HPAD, HPAD, D_EMB);
}
